// Net_77644418777668
// MI455X (gfx1250) — matrix operand order repaired and checked by reading, not compiled, not run on hardware
//
#include <hip/hip_runtime.h>

typedef __attribute__((ext_vector_type(16))) _Float16 v16h;
typedef __attribute__((ext_vector_type(8)))  float    v8f;
typedef __attribute__((ext_vector_type(4)))  float    v4f;   // native vector: OK for nontemporal builtins

#define L_ROWS 262144
#define ROWS_PER_WAVE 16
#define WAVES_PER_BLOCK 4
#define ROWS_PER_BLOCK (ROWS_PER_WAVE * WAVES_PER_BLOCK)

__global__ __launch_bounds__(128) void ddk_mlp_fused(
    const float* __restrict__ x,      // (L,1,11,11) flat
    const float* __restrict__ alpha,  // (1,)
    const float* __restrict__ beta,   // (1,)
    const float* __restrict__ W1,     // (32,121) row-major
    const float* __restrict__ W2,     // (10,32) row-major
    float* __restrict__ out)          // (L,10)
{
    // LDS staging
    __shared__ float     xs[WAVES_PER_BLOCK * 1936];       // 16 rows x 121 per wave (reused as out-stage)
    __shared__ _Float16  ws[WAVES_PER_BLOCK * 16 * 128];   // recurrence out, K padded to 128
    __shared__ _Float16  hs[WAVES_PER_BLOCK * 16 * 32];    // sigmoid(h), 16x32 per wave
    __shared__ _Float16  w1s[32 * 128];                    // W1 in f16, K padded 121->128
    __shared__ _Float16  w2s[16 * 32];                     // W2 in f16, N padded 10->16

    const int tid  = threadIdx.x;
    const int lane = tid & 31;
    const int wave = tid >> 5;
    const int hgrp = lane >> 4;        // half-wave group (0: lanes 0-15, 1: lanes 16-31)
    const int lm   = lane & 15;
    const int rowbase = blockIdx.x * ROWS_PER_BLOCK + wave * ROWS_PER_WAVE;

    // ---- stage weights (block cooperative; W1/W2 are L2-resident) ----
    for (int i = tid; i < 32 * 128; i += 128) {
        int n = i >> 7, k = i & 127;
        w1s[i] = (k < 121) ? (_Float16)W1[n * 121 + k] : (_Float16)0.0f;
    }
    for (int i = tid; i < 16 * 32; i += 128) {
        int n = i >> 5, k = i & 31;
        w2s[i] = (n < 10) ? (_Float16)W2[n * 32 + k] : (_Float16)0.0f;
    }

    // ---- stage x: 16 rows x 121 f32 = 484 x 128-bit per wave, coalesced + non-temporal ----
    {
        const v4f* xg = (const v4f*)(x + (size_t)rowbase * 121);   // 7744B-aligned region
        v4f* xl = (v4f*)(xs + wave * 1936);
        for (int i = lane; i < 484; i += 32) xl[i] = __builtin_nontemporal_load(xg + i);
    }
    __syncthreads();

    const float a = alpha[0];
    const float b = beta[0];

    _Float16* wsw = ws + wave * (16 * 128);
    const float* xw = xs + wave * 1936;

    // zero the K-pad (cols 121..127) so WMMA K-padding contributes 0
    for (int i = lane; i < 16 * 7; i += 32) {
        int r = i / 7, c = 121 + (i % 7);
        wsw[r * 128 + c] = (_Float16)0.0f;
    }

    // ---- recurrence: 176 chains (16 rows x 11 m-rows), length 11 each ----
    for (int c = lane; c < 176; c += 32) {
        int r = c / 11, m = c % 11;
        const float* xp = xw + r * 121 + m * 11;
        _Float16* wp = wsw + r * 128 + m * 11;
        float w = 1.0f;
        wp[0] = (_Float16)1.0f;
#pragma unroll
        for (int n = 1; n < 11; ++n) {
            w = w + a / w - b * xp[n - 1];
            wp[n] = (_Float16)w;
        }
    }
    __syncthreads();

    // ---- GEMM1: (16x128) @ (128x32) = 2 N-tiles x 4 K-steps of wmma f16 ----
    v8f c0 = {};
    v8f c1 = {};
#pragma unroll
    for (int kk = 0; kk < 4; ++kk) {
        const int kbase = kk * 32;
        // A fragment (ISA 16-bit A 16x32 layout): lane = row M=lm,
        // K = kbase + e + (e>=8 ? 8 : 0) + hgrp*8  -> two contiguous 8-half runs
        v16h afrag;
#pragma unroll
        for (int e = 0; e < 16; ++e) {
            int K = kbase + e + ((e >= 8) ? 8 : 0) + hgrp * 8;
            afrag[e] = wsw[lm * 128 + K];
        }
        // B fragments: lane = col N=lm(+16); K = kbase + e + (e>=8 ? 8 : 0) + hgrp*8 (same as A)
        const int kb = kbase + hgrp * 8;
        v16h b0, b1;
#pragma unroll
        for (int e = 0; e < 16; ++e) {
            b0[e] = w1s[lm * 128 + kb + e + ((e >= 8) ? 8 : 0)];
            b1[e] = w1s[(lm + 16) * 128 + kb + e + ((e >= 8) ? 8 : 0)];
        }
        c0 = __builtin_amdgcn_wmma_f32_16x16x32_f16(false, afrag, false, b0,
                                                    (short)0, c0, false, false);
        c1 = __builtin_amdgcn_wmma_f32_16x16x32_f16(false, afrag, false, b1,
                                                    (short)0, c1, false, false);
    }

    // ---- sigmoid + transpose h into LDS (C layout != A layout) ----
    _Float16* hw = hs + wave * (16 * 32);
#pragma unroll
    for (int j = 0; j < 8; ++j) {
        int M = j + 8 * hgrp;
        float s0 = 1.0f / (1.0f + __expf(-c0[j]));
        float s1 = 1.0f / (1.0f + __expf(-c1[j]));
        hw[M * 32 + lm]      = (_Float16)s0;
        hw[M * 32 + lm + 16] = (_Float16)s1;
    }
    __syncthreads();

    // ---- GEMM2: (16x32) @ (32x16), one wmma; cols 10..15 of B are zero ----
    v16h a2, b2;
#pragma unroll
    for (int e = 0; e < 16; ++e) {
        int K = e + ((e >= 8) ? 8 : 0) + hgrp * 8;
        a2[e] = hw[lm * 32 + K];
        b2[e] = w2s[lm * 32 + K];
    }
    v8f c2 = {};
    c2 = __builtin_amdgcn_wmma_f32_16x16x32_f16(false, a2, false, b2,
                                                (short)0, c2, false, false);

    // ---- stage D2 through LDS (reuse xs) and store coalesced 128-bit, NT ----
    float* obuf = xs + wave * 1936;  // need 160 floats; xs is dead after GEMM1
    if (lm < 10) {
#pragma unroll
        for (int j = 0; j < 8; ++j) {
            int M = j + 8 * hgrp;
            obuf[M * 10 + lm] = c2[j];
        }
    }
    __syncthreads();  // uniform; orders the wave-local LDS transpose

    {
        v4f* og = (v4f*)(out + (size_t)rowbase * 10);   // 640B-aligned
        const v4f* ol = (const v4f*)obuf;
        for (int i = lane; i < 40; i += 32)
            __builtin_nontemporal_store(ol[i], og + i);
    }
}

extern "C" void kernel_launch(void* const* d_in, const int* in_sizes, int n_in,
                              void* d_out, int out_size, void* d_ws, size_t ws_size,
                              hipStream_t stream) {
    (void)in_sizes; (void)n_in; (void)d_ws; (void)ws_size; (void)out_size;
    const float* x     = (const float*)d_in[0];
    const float* alpha = (const float*)d_in[1];
    const float* beta  = (const float*)d_in[2];
    const float* W1    = (const float*)d_in[3];
    const float* W2    = (const float*)d_in[4];
    float* out = (float*)d_out;

    dim3 grid(L_ROWS / ROWS_PER_BLOCK);  // 4096
    dim3 block(128);                     // 4 wave32 per block
    ddk_mlp_fused<<<grid, block, 0, stream>>>(x, alpha, beta, W1, W2, out);
}
